// binary_decoder_15771119911214
// MI455X (gfx1250) — matrix operand order repaired and checked by reading, not compiled, not run on hardware
//
#include <hip/hip_runtime.h>
#include <hip/hip_bf16.h>

// Problem constants (from reference)
#define IN_F   4096
#define OUT_F  1024
#define NBITS  8
#define OUT_B  (OUT_F * NBITS)   // 8192 columns of the binarized GEMV
#define KSEG   8                 // K-split: 4096 waves -> ~16 MB loads in flight
#define KLEN   (IN_F / KSEG)     // 512

typedef __attribute__((ext_vector_type(16))) _Float16 v16h;
typedef __attribute__((ext_vector_type(8)))  _Float16 v8h;
typedef __attribute__((ext_vector_type(8)))  float    v8f;

// ---------------------------------------------------------------------------
// Kernel 1: S_part[seg][n] = sum_{k in seg} latent[k] * (weight[k][n] > 0.5)
// via v_wmma_f32_16x16x32_f16. Each wave owns a 16-column strip and one of
// KSEG K-segments; all 16 A rows carry the same latent chunk so D row 0 is
// the GEMV result. Weight (134 MB, touched once) streams with NT loads.
// ---------------------------------------------------------------------------
__global__ void __launch_bounds__(128)
gemv_bin_wmma(const float* __restrict__ latent,
              const float* __restrict__ weight,
              float* __restrict__ S_part)
{
    __shared__ __align__(16) _Float16 lat[KLEN];   // 1 KB: latent segment as f16

    const int tid   = threadIdx.x;                 // 0..127 (4 waves)
    const int kseg  = blockIdx.y;
    const int kbase = kseg * KLEN;

    // Stage this block's latent segment into LDS (f32 -> f16 once per block).
    for (int i = tid; i < KLEN; i += 128)
        lat[i] = (_Float16)latent[kbase + i];
    __syncthreads();

    const int wave = tid >> 5;
    const int lane = tid & 31;
    const int n0   = (blockIdx.x * 4 + wave) * 16; // 16-column strip
    const int col  = n0 + (lane & 15);             // B: lane holds this column
    const int kh   = lane >> 4;                    // which half-wave K group

    v8f acc = {};

    #pragma unroll 2
    for (int kb = 0; kb < KLEN; kb += 32) {
        // ---- A tile (16x32 f16): every row m = latent[kbase+kb .. +31].
        // Layout (matches B): lanes 0-15 hold K {0..15}; lanes 16-31 hold
        // K {16..31}; two halves per VGPR.
        const _Float16* lp = &lat[kb + kh * 16];
        v8h alo = *(const v8h*)(lp);        // K = kh*16 + 0..7     (VGPR 0..3)
        v8h ahi = *(const v8h*)(lp + 8);    // K = kh*16 + 8..15    (VGPR 4..7)
        v16h a  = __builtin_shufflevector(alo, ahi,
                    0,1,2,3,4,5,6,7, 8,9,10,11,12,13,14,15);

        // ---- B tile (32x16 f16): binarize on the fly. Lane holds column
        // `col`; its 16 K rows are kbase+kb+kh*16 .. +15 (pairs per VGPR).
        const float* wp = weight + (size_t)(kbase + kb + kh * 16) * OUT_B + col;
        v16h b;
        #pragma unroll
        for (int j = 0; j < 16; ++j) {
            float w = __builtin_nontemporal_load(wp + (size_t)j * OUT_B);
            b[j] = (w > 0.5f) ? (_Float16)1.0f : (_Float16)0.0f;
        }

        acc = __builtin_amdgcn_wmma_f32_16x16x32_f16(
                  /*neg_a=*/false, a, /*neg_b=*/false, b,
                  /*c_mod=*/(short)0, acc,
                  /*reuse_a=*/false, /*reuse_b=*/false);
    }

    // D row 0 lives in acc[0] on lanes 0..15 (N = lane).
    if (lane < 16)
        S_part[(size_t)kseg * OUT_B + n0 + lane] = acc[0];
}

// ---------------------------------------------------------------------------
// Kernel 2: one thread per neuron. Sum K-partials, closed-form ripple-carry
// (floor carries telescope, so acc_b = T_b mod 2 with c_{b+1} = floor(T_b/2)),
// pred/int_s via two's-complement powers, block-wide loss reduction.
// ---------------------------------------------------------------------------
__global__ void __launch_bounds__(1024)
finalize_kernel(const float* __restrict__ S_part,
                const float* __restrict__ true_sum,
                float* __restrict__ out)
{
    __shared__ float red[1024];
    const int o = threadIdx.x;   // neuron 0..1023

    float s[NBITS], ts[NBITS];
    #pragma unroll
    for (int b = 0; b < NBITS; ++b) {
        float v = 0.0f;
        #pragma unroll
        for (int seg = 0; seg < KSEG; ++seg)
            v += S_part[(size_t)seg * OUT_B + o * NBITS + b];
        s[b]  = v;
        ts[b] = true_sum[o * NBITS + b];
    }

    const float powers[NBITS] = {1.f, 2.f, 4.f, 8.f, 16.f, 32.f, 64.f, -128.f};
    float pred = 0.0f, ints = 0.0f;
    #pragma unroll
    for (int b = 0; b < NBITS; ++b) {
        pred += powers[b] * s[b];
        ints += powers[b] * ts[b];
    }

    // Ripple-carry closed form -> trigger = acc - true_sum_bits
    float c = 0.0f;
    #pragma unroll
    for (int b = 0; b < NBITS; ++b) {
        float T = s[b] + c;
        c = floorf(T * 0.5f);
        out[1 + o * NBITS + b] = (T - 2.0f * c) - ts[b];
    }

    // loss = 0.5 * mean((pred - int_s)^2) over 1024 neurons
    float d = pred - ints;
    red[o] = d * d;
    __syncthreads();
    #pragma unroll
    for (int stride = 512; stride > 0; stride >>= 1) {
        if (o < stride) red[o] += red[o + stride];
        __syncthreads();
    }
    if (o == 0) out[0] = 0.5f * red[0] * (1.0f / 1024.0f);
}

// ---------------------------------------------------------------------------
extern "C" void kernel_launch(void* const* d_in, const int* in_sizes, int n_in,
                              void* d_out, int out_size, void* d_ws, size_t ws_size,
                              hipStream_t stream)
{
    const float* latent   = (const float*)d_in[0];  // [1, 4096]
    const float* true_sum = (const float*)d_in[1];  // [1, 8192]
    const float* weight   = (const float*)d_in[2];  // [4096, 8192]
    float* out = (float*)d_out;                     // [1 + 8192]
    float* S_part = (float*)d_ws;                   // KSEG * 8192 floats (256 KB)

    dim3 grid(OUT_B / (16 * 4), KSEG);              // 128 column-blocks x 8 K-segments
    gemv_bin_wmma<<<grid, 128, 0, stream>>>(latent, weight, S_part);
    finalize_kernel<<<1, 1024, 0, stream>>>(S_part, true_sum, out);
}
